// Attention_8735963480466
// MI455X (gfx1250) — matrix operand order repaired and checked by reading, not compiled, not run on hardware
//
#include <hip/hip_runtime.h>

typedef unsigned short u16;
typedef unsigned int   u32;
typedef __attribute__((ext_vector_type(16))) __bf16 v16bf;
typedef __attribute__((ext_vector_type(2)))  __bf16 v2bf;
typedef __attribute__((ext_vector_type(8)))  float  v8f;

#define DIM   1024
#define NH    16
#define HD    64
#define BATCH 4
#define SEQ   2048
#define NROW  (BATCH * SEQ)   // 8192
#define SCALE 0.125f

union BFrag { v16bf v; uint4 u[2]; };

// hardware bf16 convert (fptrunc -> v_cvt bf16 path on gfx1250)
__device__ __forceinline__ u16 f2bf(float f) {
  return __builtin_bit_cast(u16, (__bf16)f);
}
__device__ __forceinline__ u32 pack_bf16(float a, float b) {
  v2bf p = { (__bf16)a, (__bf16)b };   // should select v_cvt_pk_bf16_f32
  return __builtin_bit_cast(u32, p);
}
__device__ __forceinline__ void wait_lds() {
  asm volatile("s_wait_dscnt 0" ::: "memory");
}
__device__ __forceinline__ v8f wmma_bf16(const BFrag& a, const BFrag& b, v8f c) {
  return __builtin_amdgcn_wmma_f32_16x16x32_bf16(false, a.v, false, b.v,
                                                 (short)0, c, false, false);
}

// ---------------------------------------------------------------------------
// Kernel 1: qkv = x @ w_qkv  (f32 in, bf16 out). Q,K row-major; V transposed.
// block = 256 thr (8 waves), tile 128(M) x 128(N), K stepped by 32.
// Wave grid 4(M) x 2(N); each wave owns 2x4 16x16 C tiles -> 8 WMMA / step.
// ---------------------------------------------------------------------------
__global__ __launch_bounds__(256) void qkv_gemm_8735963480466(
    const float* __restrict__ x, const float* __restrict__ wqkv,
    u16* __restrict__ qb, u16* __restrict__ kbuf, u16* __restrict__ vtb) {
  __shared__ u16 As[128 * 48];   // A tile, row stride 48 (16B aligned rows)
  __shared__ u16 Bt[128 * 32];   // B tile transposed: Bt[n][k]
  __shared__ u16 Ep[8 * 256];    // per-wave epilogue transpose buffer

  const int t = threadIdx.x;
  const int wave = t >> 5, lane = t & 31, hi = lane >> 4, ln = lane & 15;
  const int wm = wave >> 1, wn = wave & 1;
  const int m0b = blockIdx.x * 128, n0b = blockIdx.y * 128;

  v8f acc[2][4] = {};

  for (int ks = 0; ks < DIM; ks += 32) {
#pragma unroll
    for (int it = 0; it < 4; ++it) {               // stage A: 128x32 f32->bf16
      int idx = t + it * 256, row = idx >> 3, c4 = idx & 7;
      float4 f = *(const float4*)(x + (size_t)(m0b + row) * DIM + ks + c4 * 4);
      *(uint2*)&As[row * 48 + c4 * 4] =
          make_uint2(pack_bf16(f.x, f.y), pack_bf16(f.z, f.w));
    }
#pragma unroll
    for (int it = 0; it < 4; ++it) {               // stage B transposed 32x128
      int idx = t + it * 256, row = idx >> 5, c4 = idx & 31;
      float4 f = *(const float4*)(wqkv + (size_t)(ks + row) * (3 * DIM) + n0b + c4 * 4);
      Bt[(c4 * 4 + 0) * 32 + row] = f2bf(f.x);
      Bt[(c4 * 4 + 1) * 32 + row] = f2bf(f.y);
      Bt[(c4 * 4 + 2) * 32 + row] = f2bf(f.z);
      Bt[(c4 * 4 + 3) * 32 + row] = f2bf(f.w);
    }
    __syncthreads();

    BFrag a[2], b[4];
#pragma unroll
    for (int i = 0; i < 2; ++i) {                  // A 16x32 bf16 lane layout
      int row = wm * 32 + i * 16 + ln;
      a[i].u[0] = *(const uint4*)&As[row * 48 + hi * 8];
      a[i].u[1] = *(const uint4*)&As[row * 48 + 16 + hi * 8];
    }
#pragma unroll
    for (int j = 0; j < 4; ++j) {                  // B 32x16 bf16 lane layout
      int n = wn * 64 + j * 16 + ln;
      b[j].u[0] = *(const uint4*)&Bt[n * 32 + hi * 8];
      b[j].u[1] = *(const uint4*)&Bt[n * 32 + 16 + hi * 8];
    }
#pragma unroll
    for (int i = 0; i < 2; ++i)
#pragma unroll
      for (int j = 0; j < 4; ++j)
        acc[i][j] = wmma_bf16(a[i], b[j], acc[i][j]);
    __syncthreads();
  }

  // epilogue: route by qkv column region
  u16* ep = &Ep[wave * 256];
#pragma unroll
  for (int i = 0; i < 2; ++i) {
#pragma unroll
    for (int j = 0; j < 4; ++j) {
      int m0 = m0b + wm * 32 + i * 16;
      int n0 = n0b + wn * 64 + j * 16;
      int region = n0 >> 10;                       // 0=Q 1=K 2=V
      if (region < 2) {
        // LDS transpose C-layout -> row-major, coalesced b128 store
#pragma unroll
        for (int r = 0; r < 8; ++r)
          ep[(hi * 8 + r) * 16 + ln] = f2bf(acc[i][j][r]);
        wait_lds();
        uint4 vrow = *(const uint4*)&ep[ln * 16 + hi * 8];
        u16* dst = region ? kbuf : qb;
        *(uint4*)&dst[(size_t)(m0 + ln) * DIM + (n0 & 1023) + hi * 8] = vrow;
        wait_lds();
      } else {
        // transposed V store: 8 consecutive s per lane -> one 16B store
        int nl = n0 - 2 * DIM;
        int hh = nl >> 6, d = (nl & 63) + ln;
        int bb = m0 >> 11, s0 = (m0 & 2047) + hi * 8;
        uint4 pk;
        pk.x = pack_bf16(acc[i][j][0], acc[i][j][1]);
        pk.y = pack_bf16(acc[i][j][2], acc[i][j][3]);
        pk.z = pack_bf16(acc[i][j][4], acc[i][j][5]);
        pk.w = pack_bf16(acc[i][j][6], acc[i][j][7]);
        *(uint4*)&vtb[(((size_t)(bb * NH + hh)) * HD + d) * SEQ + s0] = pk;
      }
    }
  }
}

// ---------------------------------------------------------------------------
// Kernel 2: flash attention. block = 128 thr (4 waves), each wave = 16 q rows
// of one (b,h); keys streamed in 32-key steps with online softmax.
// ---------------------------------------------------------------------------
__global__ __launch_bounds__(128) void attn_8735963480466(
    const u16* __restrict__ qb, const u16* __restrict__ kbuf,
    const u16* __restrict__ vtb, u16* __restrict__ ctx) {
  __shared__ u16 Pb[4 * 512];    // per-wave P tile 16x32 bf16
  __shared__ u16 Cb[4 * 1024];   // per-wave ctx tile 16x64 bf16

  const int t = threadIdx.x;
  const int wave = t >> 5, lane = t & 31, hi = lane >> 4, ln = lane & 15;
  const int bh = blockIdx.x, b = bh >> 4, h = bh & 15;
  const int qrow0 = b * SEQ + blockIdx.y * 64 + wave * 16;
  const int hb = h * HD;
  const size_t vbase = (size_t)(b * NH + h) * HD;

  BFrag q[2];
#pragma unroll
  for (int c = 0; c < 2; ++c) {                    // Q fragments: registers
    const u16* qp = qb + (size_t)(qrow0 + ln) * DIM + hb + c * 32 + hi * 8;
    q[c].u[0] = *(const uint4*)qp;
    q[c].u[1] = *(const uint4*)(qp + 16);
  }

  float mrun[8], lrun[8];
#pragma unroll
  for (int r = 0; r < 8; ++r) { mrun[r] = -3.0e38f; lrun[r] = 0.f; }
  v8f acc[4] = {};

  for (int kb0 = 0; kb0 < SEQ; kb0 += 32) {
    if (kb0 + 32 < SEQ)
      __builtin_prefetch(kbuf + (size_t)(b * SEQ + kb0 + 32 + lane) * DIM + hb, 0, 1);

    v8f s[2] = {};
#pragma unroll
    for (int tt = 0; tt < 2; ++tt) {               // scores for 2x16 keys
#pragma unroll
      for (int c = 0; c < 2; ++c) {
        BFrag kf;
        const u16* kp = kbuf + (size_t)(b * SEQ + kb0 + tt * 16 + ln) * DIM
                             + hb + c * 32 + hi * 8;
        kf.u[0] = *(const uint4*)kp;
        kf.u[1] = *(const uint4*)(kp + 16);
        s[tt] = wmma_bf16(q[c], kf, s[tt]);
      }
    }

    float p0[8], p1[8], corr[8];
#pragma unroll
    for (int r = 0; r < 8; ++r) {                  // online softmax per row
      float a0 = s[0][r] * SCALE, a1 = s[1][r] * SCALE;
      float bm = fmaxf(a0, a1);
#pragma unroll
      for (int off = 1; off < 16; off <<= 1)
        bm = fmaxf(bm, __shfl_xor(bm, off, 32));
      float mn = fmaxf(mrun[r], bm);
      corr[r] = __expf(mrun[r] - mn);
      p0[r] = __expf(a0 - mn);
      p1[r] = __expf(a1 - mn);
      float rs = p0[r] + p1[r];
#pragma unroll
      for (int off = 1; off < 16; off <<= 1)
        rs += __shfl_xor(rs, off, 32);
      lrun[r] = lrun[r] * corr[r] + rs;
      mrun[r] = mn;
    }
#pragma unroll
    for (int nt = 0; nt < 4; ++nt)
#pragma unroll
      for (int r = 0; r < 8; ++r)
        acc[nt][r] *= corr[r];

    // P: C-layout -> A-layout through per-wave LDS
    u16* Pw = &Pb[wave * 512];
#pragma unroll
    for (int r = 0; r < 8; ++r) {
      Pw[(hi * 8 + r) * 32 + ln]      = f2bf(p0[r]);
      Pw[(hi * 8 + r) * 32 + 16 + ln] = f2bf(p1[r]);
    }
    wait_lds();
    BFrag pa;
    pa.u[0] = *(const uint4*)&Pw[ln * 32 + hi * 8];
    pa.u[1] = *(const uint4*)&Pw[ln * 32 + 16 + hi * 8];

#pragma unroll
    for (int nt = 0; nt < 4; ++nt) {               // P @ V (transposed V)
      BFrag vf;
      const u16* vp = vtb + (vbase + nt * 16 + ln) * SEQ + kb0 + hi * 8;
      vf.u[0] = *(const uint4*)vp;
      vf.u[1] = *(const uint4*)(vp + 16);
      acc[nt] = wmma_bf16(pa, vf, acc[nt]);
    }
  }

  // normalize + coalesced store via per-wave LDS transpose
  u16* Cw = &Cb[wave * 1024];
#pragma unroll
  for (int r = 0; r < 8; ++r) {
    float inv = 1.f / lrun[r];
#pragma unroll
    for (int nt = 0; nt < 4; ++nt)
      Cw[(hi * 8 + r) * 64 + nt * 16 + ln] = f2bf(acc[nt][r] * inv);
  }
  wait_lds();
  int row = lane >> 1, half = lane & 1;
#pragma unroll
  for (int qq = 0; qq < 4; ++qq) {
    uint4 v = *(const uint4*)&Cw[row * 64 + half * 32 + qq * 8];
    *(uint4*)&ctx[(size_t)(qrow0 + row) * DIM + hb + half * 32 + qq * 8] = v;
  }
}

// ---------------------------------------------------------------------------
// Kernel 3: out = ctx @ w_out + b_out (bf16 A, f32 weights converted, f32 out)
// block tile 128x128, wave tile 32x64 (2x4 WMMA tiles).
// ---------------------------------------------------------------------------
__global__ __launch_bounds__(256) void out_gemm_8735963480466(
    const u16* __restrict__ ctx, const float* __restrict__ wout,
    const float* __restrict__ bout, float* __restrict__ out) {
  __shared__ u16   As[128 * 48];
  __shared__ u16   Bt[128 * 32];
  __shared__ float Ef[8 * 256];

  const int t = threadIdx.x;
  const int wave = t >> 5, lane = t & 31, hi = lane >> 4, ln = lane & 15;
  const int wm = wave >> 1, wn = wave & 1;
  const int m0b = blockIdx.x * 128, n0b = blockIdx.y * 128;

  v8f acc[2][4] = {};

  for (int ks = 0; ks < DIM; ks += 32) {
#pragma unroll
    for (int it = 0; it < 2; ++it) {               // stage A (already bf16)
      int idx = t + it * 256, row = idx >> 2, c4 = idx & 3;
      uint4 v = *(const uint4*)(ctx + (size_t)(m0b + row) * DIM + ks + c4 * 8);
      *(uint4*)&As[row * 48 + c4 * 8] = v;
    }
#pragma unroll
    for (int it = 0; it < 4; ++it) {               // stage B transposed 32x128
      int idx = t + it * 256, row = idx >> 5, c4 = idx & 31;
      float4 f = *(const float4*)(wout + (size_t)(ks + row) * DIM + n0b + c4 * 4);
      Bt[(c4 * 4 + 0) * 32 + row] = f2bf(f.x);
      Bt[(c4 * 4 + 1) * 32 + row] = f2bf(f.y);
      Bt[(c4 * 4 + 2) * 32 + row] = f2bf(f.z);
      Bt[(c4 * 4 + 3) * 32 + row] = f2bf(f.w);
    }
    __syncthreads();

    BFrag a[2], b[4];
#pragma unroll
    for (int i = 0; i < 2; ++i) {
      int row = wm * 32 + i * 16 + ln;
      a[i].u[0] = *(const uint4*)&As[row * 48 + hi * 8];
      a[i].u[1] = *(const uint4*)&As[row * 48 + 16 + hi * 8];
    }
#pragma unroll
    for (int j = 0; j < 4; ++j) {
      int n = wn * 64 + j * 16 + ln;
      b[j].u[0] = *(const uint4*)&Bt[n * 32 + hi * 8];
      b[j].u[1] = *(const uint4*)&Bt[n * 32 + 16 + hi * 8];
    }
#pragma unroll
    for (int i = 0; i < 2; ++i)
#pragma unroll
      for (int j = 0; j < 4; ++j)
        acc[i][j] = wmma_bf16(a[i], b[j], acc[i][j]);
    __syncthreads();
  }

  float* ef = &Ef[wave * 256];
#pragma unroll
  for (int i = 0; i < 2; ++i) {
#pragma unroll
    for (int j = 0; j < 4; ++j) {
      int m0 = m0b + wm * 32 + i * 16;
      int n0 = n0b + wn * 64 + j * 16;
      float bv = bout[n0 + ln];                    // bias: one col per lane
#pragma unroll
      for (int r = 0; r < 8; ++r)
        ef[(hi * 8 + r) * 16 + ln] = acc[i][j][r] + bv;
      wait_lds();
      float4 f0 = *(const float4*)&ef[ln * 16 + hi * 8];
      float4 f1 = *(const float4*)&ef[ln * 16 + hi * 8 + 4];
      float* dst = out + (size_t)(m0 + ln) * DIM + n0 + hi * 8;
      *(float4*)dst       = f0;
      *(float4*)(dst + 4) = f1;
      wait_lds();
    }
  }
}

// ---------------------------------------------------------------------------
extern "C" void kernel_launch(void* const* d_in, const int* in_sizes, int n_in,
                              void* d_out, int out_size, void* d_ws, size_t ws_size,
                              hipStream_t stream) {
  (void)in_sizes; (void)n_in; (void)out_size; (void)ws_size;
  const float* x    = (const float*)d_in[0];   // [4,2048,1024]
  const float* wqkv = (const float*)d_in[1];   // [1024,3072]
  const float* wout = (const float*)d_in[2];   // [1024,1024]
  const float* bout = (const float*)d_in[3];   // [1024]
  float* out = (float*)d_out;                  // [4,2048,1024]

  const size_t buf = (size_t)NROW * DIM;       // 8M bf16 elems each
  u16* qb   = (u16*)d_ws;                      // Q  [8192,1024] bf16
  u16* kbuf = qb + buf;                        // K  [8192,1024] bf16
  u16* vtb  = kbuf + buf;                      // Vt [B,H,64,2048] bf16
  u16* ctx  = vtb + buf;                       // ctx[8192,1024] bf16

  qkv_gemm_8735963480466<<<dim3(NROW / 128, (3 * DIM) / 128), 256, 0, stream>>>(
      x, wqkv, qb, kbuf, vtb);
  attn_8735963480466<<<dim3(BATCH * NH, SEQ / 64), 128, 0, stream>>>(
      qb, kbuf, vtb, ctx);
  out_gemm_8735963480466<<<dim3(NROW / 128, DIM / 128), 256, 0, stream>>>(
      ctx, wout, bout, out);
}
